// RNN_13013750907457
// MI455X (gfx1250) — matrix operand order repaired and checked by reading, not compiled, not run on hardware
//
#include <hip/hip_runtime.h>
#include <hip/hip_bf16.h>

typedef __attribute__((ext_vector_type(16))) __bf16 v16bf;
typedef __attribute__((ext_vector_type(8)))  __bf16 v8bf;
typedef __attribute__((ext_vector_type(8)))  float  v8f;

#define T_SEQ 256
#define BATCH 64
#define HID   1024
#define OUTD  10000

// GEMM block staging geometry
#define KC         64   // K elements staged per chunk
#define BCOLS      64   // output columns staged per block
#define LDS_STRIDE 72   // padded row stride in elements (144 B) to spread banks

#define WMMA_BF16(a, b, c) \
  __builtin_amdgcn_wmma_f32_16x16x32_bf16(false, (a), false, (b), (short)0, (c), false, false)

__device__ __forceinline__ __bf16 f2bf(float x) {
  unsigned u = __builtin_bit_cast(unsigned, x);
  u += 0x7FFFu + ((u >> 16) & 1u);           // round-to-nearest-even
  unsigned short s = (unsigned short)(u >> 16);
  return __builtin_bit_cast(__bf16, s);
}

__device__ __forceinline__ float sigm(float x) { return 1.0f / (1.0f + __expf(-x)); }

// Generic shared pointers carry the LDS byte offset in their low 32 bits
__device__ __forceinline__ unsigned lds_off_of(const void* p) {
  return (unsigned)(unsigned long long)p;
}

// A fragment: 16x32 bf16 tile, row-major A (lda in elements).
// Lanes 0-15: row=lane, halves 0..15 = K k0..k0+15.
// Lanes 16-31: row=lane-16, halves 0..15 = K k0+16..k0+31.
__device__ __forceinline__ v16bf load_a_frag(const __bf16* A, size_t lda, int row0, int k0, int lane) {
  int r  = lane & 15;
  int kb = (lane >> 4) << 4;
  const __bf16* p = A + (size_t)(row0 + r) * lda + (size_t)(k0 + kb);
  v8bf lo = *(const v8bf*)(p);
  v8bf hi = *(const v8bf*)(p + 8);
  return __builtin_shufflevector(lo, hi, 0,1,2,3,4,5,6,7,8,9,10,11,12,13,14,15);
}

// B fragment straight from N-major global storage BT[n][k] (used by lstm_step).
__device__ __forceinline__ v16bf load_b_frag(const __bf16* BT, size_t ldbt, int col0, int k0, int lane) {
  int n  = lane & 15;
  int kb = (lane >> 4) << 4;
  const __bf16* p = BT + (size_t)(col0 + n) * ldbt + (size_t)(k0 + kb);
  return *(const v16bf*)(p);
}

// B fragment from the LDS-staged panel (padded stride, two 16B loads).
__device__ __forceinline__ v16bf load_b_lds(const __bf16* buf, int col_local, int klocal, int lane) {
  int n  = lane & 15;
  int kb = (lane >> 4) << 4;
  const __bf16* p = buf + (size_t)(col_local + n) * LDS_STRIDE + (size_t)(klocal + kb);
  v8bf lo = *(const v8bf*)(p);
  v8bf hi = *(const v8bf*)(p + 8);
  return __builtin_shufflevector(lo, hi, 0,1,2,3,4,5,6,7,8,9,10,11,12,13,14,15);
}

// Async DMA a BCOLS x KC bf16 panel (row n = output column) into LDS.
// Panel row = KC*2 = 128 B; 512 x 16B segments; 2 per thread via
// global_load_async_to_lds_b128 (tracked by ASYNCcnt, no VGPR round trip).
__device__ __forceinline__ void stage_b_async(const __bf16* BT, int K, int n0, int k0,
                                              __bf16* buf, int tid, int Nvalid) {
#pragma unroll
  for (int s = 0; s < 2; ++s) {
    int seg = tid + s * 256;            // 0..511
    int row = seg >> 3;                 // 0..63  (output column)
    int off = (seg & 7) * 16;           // byte offset within 128B row
    int gn  = n0 + row;
    if (gn >= Nvalid) gn = Nvalid - 1;  // clamp: harmless dup read, never OOB
    unsigned long long gaddr =
        (unsigned long long)(const void*)((const char*)(BT + (size_t)gn * K + k0) + off);
    unsigned laddr = lds_off_of((const char*)buf + row * (LDS_STRIDE * 2) + off);
    asm volatile("global_load_async_to_lds_b128 %0, %1, off"
                 :: "v"(laddr), "v"(gaddr) : "memory");
  }
}

__device__ __forceinline__ void wait_async0() {
  asm volatile("s_wait_asynccnt 0x0" ::: "memory");
}

// ---- weight convert + transpose: float [K][N] row-major -> bf16 [N][K] ----
__global__ __launch_bounds__(256)
void transpose_to_bf16(const float* __restrict__ in, __bf16* __restrict__ out, int K, int N) {
  __shared__ float tile[32][33];
  const int k0 = blockIdx.y * 32;
  const int n0 = blockIdx.x * 32;
  for (int i = threadIdx.y; i < 32; i += 8) {
    int k = k0 + i, nn = n0 + threadIdx.x;
    tile[i][threadIdx.x] = (k < K && nn < N) ? in[(size_t)k * N + nn] : 0.0f;
  }
  __syncthreads();
  for (int i = threadIdx.y; i < 32; i += 8) {
    int nn = n0 + i, k = k0 + threadIdx.x;
    if (nn < N && k < K) out[(size_t)nn * K + k] = f2bf(tile[threadIdx.x][i]);
  }
}

// ---- embedding gather + bf16 convert ----
__global__ __launch_bounds__(256)
void gather_embed(const int* __restrict__ x, const float* __restrict__ emb, __bf16* __restrict__ E) {
  const int m = blockIdx.x;
  const int idx = x[m];
  const float* src = emb + (size_t)idx * HID;
  __bf16* dst = E + (size_t)m * HID;
  for (int h = threadIdx.x; h < HID; h += 256) dst[h] = f2bf(src[h]);
}

__global__ __launch_bounds__(256)
void init_state(const float* __restrict__ h_in, const float* __restrict__ c_in,
                __bf16* __restrict__ hbf0, float* __restrict__ cws, int n) {
  int i = blockIdx.x * 256 + threadIdx.x;
  if (i < n) { hbf0[i] = f2bf(h_in[i]); cws[i] = c_in[i]; }
}

__global__ __launch_bounds__(256)
void write_final(const float* __restrict__ hF32, const float* __restrict__ cws,
                 float* __restrict__ out_tail, int n) {
  int i = blockIdx.x * 256 + threadIdx.x;
  if (i < n) { out_tail[i] = hF32[i]; out_tail[n + i] = cws[i]; }
}

// ---- bf16 WMMA GEMM with async-LDS double-buffered B panel ----
// Block: 8 waves, 128 rows x 64 cols. Wave w: M-tile blockIdx.y*8+w, 4 N-tiles.
// B panel (shared by all 8 waves) is DMA'd to LDS; copy of chunk k+1 overlaps
// WMMAs on chunk k.
template <bool OUT_BF16>
__global__ __launch_bounds__(256)
void gemm_bf16(const __bf16* __restrict__ A, const __bf16* __restrict__ BT,
               const float* __restrict__ bias, void* __restrict__ Cout,
               int N, int K, int ldc) {
  __shared__ __align__(16) __bf16 smem[2][BCOLS * LDS_STRIDE];
  const int tid  = threadIdx.x;
  const int lane = tid & 31;
  const int wave = tid >> 5;
  const int row0 = (blockIdx.y * 8 + wave) * 16;
  const int n0   = blockIdx.x * BCOLS;
  const int n    = lane & 15;
  const int rb   = (lane >> 4) << 3;

  v8f acc[4];
#pragma unroll
  for (int j = 0; j < 4; ++j) {
    const int c0 = n0 + j * 16;
    float bv = (c0 < N) ? bias[c0 + n] : 0.0f;
#pragma unroll
    for (int r = 0; r < 8; ++r) acc[j][r] = bv;
  }

  stage_b_async(BT, K, n0, 0, smem[0], tid, N);
  wait_async0();
  __syncthreads();

  int buf = 0;
  for (int kc = 0; kc < K; kc += KC, buf ^= 1) {
    if (kc + KC < K) stage_b_async(BT, K, n0, kc + KC, smem[buf ^ 1], tid, N);

#pragma unroll
    for (int kk = 0; kk < KC; kk += 32) {
      v16bf a = load_a_frag(A, (size_t)K, row0, kc + kk, lane);
#pragma unroll
      for (int j = 0; j < 4; ++j) {
        if (n0 + j * 16 < N) {
          v16bf b = load_b_lds(smem[buf], j * 16, kk, lane);
          acc[j] = WMMA_BF16(a, b, acc[j]);
        }
      }
    }
    wait_async0();       // next panel landed
    __syncthreads();     // and everyone is done reading the current one
  }

#pragma unroll
  for (int j = 0; j < 4; ++j) {
    const int c0 = n0 + j * 16;
    if (c0 >= N) continue;
#pragma unroll
    for (int r = 0; r < 8; ++r) {
      size_t idx = (size_t)(row0 + rb + r) * (size_t)ldc + (size_t)(c0 + n);
      if (OUT_BF16) ((__bf16*)Cout)[idx] = f2bf(acc[j][r]);
      else          ((float*)Cout)[idx]  = acc[j][r];
    }
  }
}

// ---- fused LSTM step: gates = Xc_t + h_in @ W_h; cell update in-register ----
__global__ __launch_bounds__(256)
void lstm_step(const __bf16* __restrict__ h_in, __bf16* __restrict__ h_out,
               const __bf16* __restrict__ WhT, const float* __restrict__ Xc_t,
               float* __restrict__ c, float* __restrict__ hf32,
               __bf16* __restrict__ res_t) {
  const int lane = threadIdx.x & 31;
  const int wave = threadIdx.x >> 5;
  const int wid  = blockIdx.x * 8 + wave;   // 0..255
  const int row0 = (wid >> 6) * 16;         // batch tile  (4)
  const int n0   = (wid & 63) * 16;         // hidden tile (64)
  const int n    = lane & 15;
  const int rb   = (lane >> 4) << 3;

  v8f acc[4];  // i, f, g, o accumulators, seeded from precomputed x-contribution
#pragma unroll
  for (int g = 0; g < 4; ++g) {
    const int c0 = g * HID + n0;
#pragma unroll
    for (int r = 0; r < 8; ++r)
      acc[g][r] = Xc_t[(size_t)(row0 + rb + r) * (4 * HID) + (size_t)(c0 + n)];
  }

  for (int k0 = 0; k0 < HID; k0 += 32) {
    v16bf a = load_a_frag(h_in, HID, row0, k0, lane);
#pragma unroll
    for (int g = 0; g < 4; ++g) {
      v16bf b = load_b_frag(WhT, HID, g * HID + n0, k0, lane);
      acc[g] = WMMA_BF16(a, b, acc[g]);
    }
  }

#pragma unroll
  for (int r = 0; r < 8; ++r) {
    const int row = row0 + rb + r;
    const int col = n0 + n;
    const size_t idx = (size_t)row * HID + col;
    float iv = sigm(acc[0][r]);
    float fv = sigm(acc[1][r]);
    float gv = tanhf(acc[2][r]);
    float ov = sigm(acc[3][r]);
    float cn = fv * c[idx] + iv * gv;
    float hn = ov * tanhf(cn);
    c[idx]    = cn;
    hf32[idx] = hn;
    __bf16 hb = f2bf(hn);
    h_out[idx] = hb;
    res_t[idx] = hb;
  }
}

extern "C" void kernel_launch(void* const* d_in, const int* in_sizes, int n_in,
                              void* d_out, int out_size, void* d_ws, size_t ws_size,
                              hipStream_t stream) {
  (void)in_sizes; (void)n_in; (void)out_size; (void)ws_size;
  const int*   x      = (const int*)  d_in[0];
  const float* h_in   = (const float*)d_in[1];
  const float* c_in   = (const float*)d_in[2];
  const float* emb    = (const float*)d_in[3];
  const float* W_proj = (const float*)d_in[4];
  const float* b_proj = (const float*)d_in[5];
  const float* W_x    = (const float*)d_in[6];
  const float* W_h    = (const float*)d_in[7];
  const float* b_lstm = (const float*)d_in[8];
  const float* W_out  = (const float*)d_in[9];
  const float* b_out  = (const float*)d_in[10];

  const int TB = T_SEQ * BATCH;  // 16384 rows
  char* ws = (char*)d_ws;
  size_t off = 0;
  auto take = [&](size_t bytes) -> char* {
    char* p = ws + off;
    off = (off + bytes + 255) & ~(size_t)255;
    return p;
  };
  __bf16* WprojT = (__bf16*)take((size_t)HID * HID * 2);
  __bf16* WxT    = (__bf16*)take((size_t)4 * HID * HID * 2);
  __bf16* WhT    = (__bf16*)take((size_t)4 * HID * HID * 2);
  __bf16* WoutT  = (__bf16*)take((size_t)OUTD * HID * 2);
  __bf16* Ebf    = (__bf16*)take((size_t)TB * HID * 2);
  __bf16* Pbf    = (__bf16*)take((size_t)TB * HID * 2);
  float*  Xc     = (float*) take((size_t)TB * 4 * HID * 4);
  __bf16* Resbf  = (__bf16*)take((size_t)TB * HID * 2);
  __bf16* hbf0   = (__bf16*)take((size_t)BATCH * HID * 2);
  __bf16* hbf1   = (__bf16*)take((size_t)BATCH * HID * 2);
  float*  cws    = (float*) take((size_t)BATCH * HID * 4);
  float*  hF32   = (float*) take((size_t)BATCH * HID * 4);

  dim3 tb(32, 8);
  transpose_to_bf16<<<dim3(HID / 32, HID / 32), tb, 0, stream>>>(W_proj, WprojT, HID, HID);
  transpose_to_bf16<<<dim3(4 * HID / 32, HID / 32), tb, 0, stream>>>(W_x, WxT, HID, 4 * HID);
  transpose_to_bf16<<<dim3(4 * HID / 32, HID / 32), tb, 0, stream>>>(W_h, WhT, HID, 4 * HID);
  transpose_to_bf16<<<dim3((OUTD + 31) / 32, HID / 32), tb, 0, stream>>>(W_out, WoutT, HID, OUTD);

  gather_embed<<<TB, 256, 0, stream>>>(x, emb, Ebf);
  init_state<<<(BATCH * HID + 255) / 256, 256, 0, stream>>>(h_in, c_in, hbf0, cws, BATCH * HID);

  // P = E @ W_proj + b_proj  (bf16 out, feeds W_x GEMM)
  gemm_bf16<true><<<dim3(HID / 64, TB / 128), 256, 0, stream>>>(Ebf, WprojT, b_proj, Pbf, HID, HID, HID);
  // Xc = P @ W_x + b_lstm  (f32 out; hoists the x-projection out of the scan)
  gemm_bf16<false><<<dim3(4 * HID / 64, TB / 128), 256, 0, stream>>>(Pbf, WxT, b_lstm, Xc, 4 * HID, HID, 4 * HID);

  __bf16* hping[2] = { hbf0, hbf1 };
  for (int t = 0; t < T_SEQ; ++t) {
    lstm_step<<<32, 256, 0, stream>>>(hping[t & 1], hping[(t + 1) & 1], WhT,
                                      Xc + (size_t)t * BATCH * 4 * HID, cws, hF32,
                                      Resbf + (size_t)t * BATCH * HID);
  }

  // output = Res @ W_out + b_out  (f32 straight into d_out)
  gemm_bf16<false><<<dim3((OUTD + 63) / 64, TB / 128), 256, 0, stream>>>(Resbf, WoutT, b_out,
                                                                         (float*)d_out, OUTD, HID, OUTD);
  // append h_f, c_f after the [T,B,O] output
  write_final<<<(BATCH * HID + 255) / 256, 256, 0, stream>>>(hF32, cws,
                                                             (float*)d_out + (size_t)TB * OUTD, BATCH * HID);
}
